// ParallelWaveganGenerator_80058190398001
// MI455X (gfx1250) — matrix operand order repaired and checked by reading, not compiled, not run on hardware
//
#include <hip/hip_runtime.h>
#include <hip/hip_bf16.h>
#include <math.h>

// ---------------------------------------------------------------------------
// Parallel WaveGAN generator, CDNA5 (gfx1250) wave32 + bf16 WMMA
// ---------------------------------------------------------------------------

typedef __attribute__((ext_vector_type(16))) __bf16 v16bf;
typedef __attribute__((ext_vector_type(8)))  float  v8f;
typedef int v4i32 __attribute__((vector_size(16)));

#define T_LEN   76800
#define NBATCH  4
#define RESC    64
#define GATEC   128
#define SKIPC   64
#define AUXC    80
#define AUXP    96     // aux channels padded to multiple of 32
#define NBLK    30
#define TT      64     // time positions per tile
#define NT      8      // tiles per workgroup (persistent weights)
#define NSLICE  11     // 6 conv K-slices + 3 aux + 2 skip/out
#define SLICE_E (GATEC * 32)          // elements per 128x32 weight slice
#define WPACK_E (NSLICE * SLICE_E)    // packed weight elements per block

// ---------------------------------------------------------------------------
// Optional CDNA5 async global->LDS copy path (guarded so compile never breaks)
// ---------------------------------------------------------------------------
#if defined(__gfx1250__) && \
    __has_builtin(__builtin_amdgcn_global_load_async_to_lds_b128) && \
    __has_builtin(__builtin_amdgcn_s_wait_asynccnt)
#define HAVE_ASYNC_LDS 1
#else
#define HAVE_ASYNC_LDS 0
#endif

__device__ __forceinline__ void cp16(void* dst_lds, const void* src_g) {
#if HAVE_ASYNC_LDS
  __builtin_amdgcn_global_load_async_to_lds_b128(
      (__attribute__((address_space(1))) v4i32*)(v4i32*)const_cast<void*>(src_g),
      (__attribute__((address_space(3))) v4i32*)(v4i32*)dst_lds,
      0, 0);
#else
  *(uint4*)dst_lds = *(const uint4*)src_g;
#endif
}

__device__ __forceinline__ void async_wait_all() {
#if HAVE_ASYNC_LDS
  __builtin_amdgcn_s_wait_asynccnt(0);
#endif
}

// ---------------------------------------------------------------------------
// WMMA fragment helpers (16x16x32 bf16, wave32 layouts per CDNA5 ISA 7.12.2)
// ---------------------------------------------------------------------------

__device__ __forceinline__ v16bf load_a_frag(const __bf16* __restrict__ wbuf,
                                             int mtile, int lane) {
  const int row   = lane & 15;
  const int khigh = lane >> 4;
  const int oc    = mtile * 16 + row;
  const __bf16* p1 = wbuf + oc * 32 + khigh * 8;
  const __bf16* p2 = p1 + 16;
  v16bf a;
#pragma unroll
  for (int e = 0; e < 8; ++e) { a[e] = p1[e]; a[8 + e] = p2[e]; }
  return a;
}

__device__ __forceinline__ v16bf load_b_frag(const __bf16* __restrict__ src,
                                             int stride, int tloc0, int kbase,
                                             int lane) {
  const int t  = tloc0 + (lane & 15);
  const int kb = kbase + ((lane >> 4) << 3);
  const __bf16* p = src + t * stride + kb;
  v16bf b;
#pragma unroll
  for (int e = 0; e < 8; ++e) { b[e] = p[e]; b[8 + e] = p[16 + e]; }
  return b;
}

// ---------------------------------------------------------------------------
// One-time weight pre-pack: fp32 -> bf16, swizzled into 128x32 K-slices
// slices 0..5: conv taps (tap = s>>1, K-half = s&1)
// slices 6..8: aux (padded 80 -> 96)
// slices 9..10: stacked [skip_w ; out_w], K-half = s-9
// ---------------------------------------------------------------------------

__global__ __launch_bounds__(256) void pack_weights_kernel(
    const float* __restrict__ conv_w,   // [30][128][64][3]
    const float* __restrict__ aux_w,    // [30][128][80]
    const float* __restrict__ skip_w,   // [30][64][64]
    const float* __restrict__ out_w,    // [30][64][64]
    __bf16* __restrict__ wpack) {
  const long g = (long)blockIdx.x * 256 + threadIdx.x;
  if (g >= (long)NBLK * WPACK_E) return;
  const int blk = (int)(g / WPACK_E);
  const int r   = (int)(g % WPACK_E);
  const int s   = r >> 12;
  const int oc  = (r >> 5) & (GATEC - 1);
  const int kk  = r & 31;
  float v;
  if (s < 6) {
    const int tap = s >> 1;
    const int ic  = ((s & 1) << 5) + kk;
    v = conv_w[(((size_t)blk * GATEC + oc) * 64 + ic) * 3 + tap];
  } else if (s < 9) {
    const int ic = ((s - 6) << 5) + kk;
    v = (ic < AUXC) ? aux_w[((size_t)blk * GATEC + oc) * AUXC + ic] : 0.0f;
  } else {
    const int ic = ((s - 9) << 5) + kk;
    v = (oc < 64) ? skip_w[((size_t)blk * 64 + oc) * 64 + ic]
                  : out_w[((size_t)blk * 64 + (oc - 64)) * 64 + ic];
  }
  wpack[g] = (__bf16)v;
}

// ---------------------------------------------------------------------------
// Upsampling network (tiny)
// ---------------------------------------------------------------------------

__global__ __launch_bounds__(256) void up_in_kernel(
    const float* __restrict__ c, const float* __restrict__ w,
    float* __restrict__ c1) {
  const int g = blockIdx.x * 256 + threadIdx.x;
  if (g >= NBATCH * AUXC * 300) return;
  const int tm = g % 300;
  const int oc = (g / 300) % AUXC;
  const int b  = g / (300 * AUXC);
  float acc = 0.f;
  for (int ic = 0; ic < AUXC; ++ic)
    acc += w[oc * AUXC + ic] * c[(b * AUXC + ic) * 300 + tm];
  c1[g] = acc;
}

// 9-tap conv along time, fp32 -> fp32; first stage samples input held x256.
__global__ __launch_bounds__(256) void up_conv_kernel(
    const float* __restrict__ in, float* __restrict__ outf,
    const float* __restrict__ w9, int hold) {
  const long g = (long)blockIdx.x * 256 + threadIdx.x;
  if (g >= (long)NBATCH * AUXC * T_LEN) return;
  const long t = g % T_LEN;
  const int ch = (int)((g / T_LEN) % AUXC);
  const int b  = (int)(g / ((long)T_LEN * AUXC));
  float acc = 0.f;
#pragma unroll
  for (int j = 0; j < 9; ++j) {
    const long tt = t + j - 4;
    if (tt < 0 || tt >= T_LEN) continue;
    const float v = hold ? in[((size_t)(b * AUXC + ch)) * 300 + (tt >> 8)]
                         : in[((size_t)(b * AUXC + ch)) * T_LEN + tt];
    acc += w9[j] * v;
  }
  outf[g] = acc;
}

// Last upsample stage: 9-tap conv + transpose to t-major [b][t][96] bf16,
// channels 80..95 zero-padded.
__global__ __launch_bounds__(256) void up_conv_pad_kernel(
    const float* __restrict__ in, __bf16* __restrict__ outb,
    const float* __restrict__ w9) {
  const long g = (long)blockIdx.x * 256 + threadIdx.x;
  if (g >= (long)NBATCH * T_LEN * AUXP) return;
  const int  ch = (int)(g % AUXP);
  const long t  = (g / AUXP) % T_LEN;
  const int  b  = (int)(g / ((long)AUXP * T_LEN));
  if (ch >= AUXC) { outb[g] = (__bf16)0.0f; return; }
  float acc = 0.f;
#pragma unroll
  for (int j = 0; j < 9; ++j) {
    const long tt = t + j - 4;
    if (tt < 0 || tt >= T_LEN) continue;
    acc += w9[j] * in[((size_t)(b * AUXC + ch)) * T_LEN + tt];
  }
  outb[g] = (__bf16)acc;
}

// ---------------------------------------------------------------------------
// First 1x1 conv from noise into t-major x (+ zero ch-major skip accumulator)
// ---------------------------------------------------------------------------

__global__ __launch_bounds__(256) void first_kernel(
    const float* __restrict__ noise, const float* __restrict__ fw,
    const float* __restrict__ fb, __bf16* __restrict__ x,
    float* __restrict__ skip) {
  const long g = (long)blockIdx.x * 256 + threadIdx.x;
  if (g >= (long)NBATCH * T_LEN * RESC) return;
  const int  ch = (int)(g & 63);
  const long t  = (g >> 6) % T_LEN;
  const int  b  = (int)(g / ((long)T_LEN * RESC));
  x[g] = (__bf16)(noise[(size_t)b * T_LEN + t] * fw[ch] + fb[ch]);
  skip[((size_t)(b * RESC + ch)) * T_LEN + t] = 0.f;
}

// ---------------------------------------------------------------------------
// Fused WaveNet residual block. Persistent weights in LDS, NT time tiles
// per workgroup. grid: (T_LEN/TT/NT, NBATCH), block: 256 (8 wave32).
// x, cu: t-major [b][t][ch]; skip: ch-major [b][ch][t].
// ---------------------------------------------------------------------------

__global__ __launch_bounds__(256) void wavenet_block_kernel(
    const __bf16* __restrict__ xin, __bf16* __restrict__ xout,
    const __bf16* __restrict__ cug, float* __restrict__ skipg,
    const __bf16* __restrict__ wpack_blk,   // [11][128][32] packed bf16
    const float* __restrict__ conv_b,       // [128]
    const float* __restrict__ skip_b,       // [64]
    const float* __restrict__ out_b,        // [64]
    int dil) {
  __shared__ __bf16 wfull[NSLICE * SLICE_E];  // 88 KB, persistent
  __shared__ __bf16 x_lds[3 * TT * RESC];     // 24 KB, [tap][t][ch]
  __shared__ __bf16 cu_lds[TT * AUXP];        // 12 KB, [t][ch]
  __shared__ __bf16 z_lds[TT * RESC];         //  8 KB, [t][ch]
  __shared__ __bf16 xo_lds[TT * RESC];        //  8 KB, [t][ch]
  __shared__ float  bias_lds[GATEC + 2 * RESC];

  const int tid  = threadIdx.x;
  const int lane = tid & 31;
  const int wave = tid >> 5;
  const int b    = blockIdx.y;

  // ---- persistent weight staging (contiguous 16B copies) ------------------
  {
    const uint4* gs = (const uint4*)wpack_blk;
    uint4*       ld = (uint4*)wfull;
    for (int i = tid; i < WPACK_E / 8; i += 256) cp16(&ld[i], &gs[i]);
  }
  if (tid < GATEC) bias_lds[tid] = conv_b[tid];
  if (tid < RESC) {
    bias_lds[GATEC + tid]        = skip_b[tid];
    bias_lds[GATEC + RESC + tid] = out_b[tid];
  }

  const int p  = wave & 3;           // M-tile pair (p, p+4)
  const int n0 = (wave >> 2) << 1;   // N-tiles n0, n0+1
  const v8f vzero = {0.f, 0.f, 0.f, 0.f, 0.f, 0.f, 0.f, 0.f};

#pragma unroll 1
  for (int it = 0; it < NT; ++it) {
    const long t0 = ((long)blockIdx.x * NT + it) * TT;

    // prefetch next tile's center tap
    if (it + 1 < NT && tid < 64)
      __builtin_prefetch(xin + ((size_t)(b * (long)T_LEN + t0 + TT)) * RESC +
                             (size_t)tid * 64, 0, 1);

    // ---- stage tiles into LDS (contiguous when no boundary clip) ---------
#pragma unroll
    for (int tap = 0; tap < 3; ++tap) {
      const long lo = t0 + (long)(tap - 1) * dil;
      __bf16* dst = x_lds + tap * (TT * RESC);
      if (lo >= 0 && lo + TT <= T_LEN) {
        const uint4* gs = (const uint4*)(xin + ((size_t)b * T_LEN + lo) * RESC);
        uint4*       ld = (uint4*)dst;
        for (int i = tid; i < (TT * RESC) / 8; i += 256) cp16(&ld[i], &gs[i]);
      } else {
        for (int idx = tid; idx < TT * RESC; idx += 256) {
          const int tl = idx >> 6;
          const int ch = idx & 63;
          const long t = lo + tl;
          dst[idx] = (t >= 0 && t < T_LEN)
                         ? xin[((size_t)b * T_LEN + t) * RESC + ch]
                         : (__bf16)0.0f;
        }
      }
    }
    {
      const uint4* gs = (const uint4*)(cug + ((size_t)b * T_LEN + t0) * AUXP);
      uint4*       ld = (uint4*)cu_lds;
      for (int i = tid; i < (TT * AUXP) / 8; i += 256) cp16(&ld[i], &gs[i]);
    }
    async_wait_all();
    __syncthreads();

    // ---- gate GEMM: h = Wconv (x) x + Waux @ cu (no barriers inside) -----
    v8f accA[2] = {vzero, vzero};
    v8f accB[2] = {vzero, vzero};
#pragma unroll
    for (int s = 0; s < 9; ++s) {
      const __bf16* cur = wfull + s * SLICE_E;
      const __bf16* src;
      int stride, kbase;
      if (s < 6) { src = x_lds + (s >> 1) * (TT * RESC); stride = RESC; kbase = (s & 1) << 5; }
      else       { src = cu_lds;                          stride = AUXP; kbase = (s - 6) << 5; }
      const v16bf aLo = load_a_frag(cur, p,     lane);
      const v16bf aHi = load_a_frag(cur, p + 4, lane);
#pragma unroll
      for (int j = 0; j < 2; ++j) {
        const v16bf bf = load_b_frag(src, stride, (n0 + j) * 16, kbase, lane);
        accA[j] = __builtin_amdgcn_wmma_f32_16x16x32_bf16(
            false, aLo, false, bf, (short)0, accA[j], false, false);
        accB[j] = __builtin_amdgcn_wmma_f32_16x16x32_bf16(
            false, aHi, false, bf, (short)0, accB[j], false, false);
      }
    }

    // ---- in-register gating, z -> LDS ------------------------------------
#pragma unroll
    for (int j = 0; j < 2; ++j) {
#pragma unroll
      for (int v = 0; v < 8; ++v) {
        const int row = v + ((lane >> 4) << 3);
        const int ch  = p * 16 + row;
        const int tl  = (n0 + j) * 16 + (lane & 15);
        const float xa = accA[j][v] + bias_lds[ch];
        const float xb = accB[j][v] + bias_lds[ch + 64];
        const float zt = tanhf(xa);
        const float zs = 1.0f / (1.0f + __expf(-xb));
        z_lds[tl * RESC + ch] = (__bf16)(zt * zs);
      }
    }
    __syncthreads();

    // ---- stacked [skip_w ; out_w] GEMM over z ----------------------------
    v8f accS[2] = {vzero, vzero};
    v8f accO[2] = {vzero, vzero};
#pragma unroll
    for (int kh = 0; kh < 2; ++kh) {
      const __bf16* cur = wfull + (9 + kh) * SLICE_E;
      const v16bf aS = load_a_frag(cur, p,     lane);
      const v16bf aO = load_a_frag(cur, p + 4, lane);
#pragma unroll
      for (int j = 0; j < 2; ++j) {
        const v16bf bf = load_b_frag(z_lds, RESC, (n0 + j) * 16, kh << 5, lane);
        accS[j] = __builtin_amdgcn_wmma_f32_16x16x32_bf16(
            false, aS, false, bf, (short)0, accS[j], false, false);
        accO[j] = __builtin_amdgcn_wmma_f32_16x16x32_bf16(
            false, aO, false, bf, (short)0, accO[j], false, false);
      }
    }

    // ---- epilogue: skip RMW (coalesced 64B runs) + residual via LDS ------
#pragma unroll
    for (int j = 0; j < 2; ++j) {
#pragma unroll
      for (int v = 0; v < 8; ++v) {
        const int  row = v + ((lane >> 4) << 3);
        const int  ch  = p * 16 + row;
        const int  tl  = (n0 + j) * 16 + (lane & 15);
        const size_t si = ((size_t)(b * RESC + ch)) * T_LEN + (t0 + tl);
        skipg[si] += accS[j][v] + bias_lds[GATEC + ch];
        const float ov = accO[j][v] + bias_lds[GATEC + RESC + ch];
        const float xc = (float)x_lds[(TT * RESC) + tl * RESC + ch]; // center
        xo_lds[tl * RESC + ch] = (__bf16)((ov + xc) * 0.25f);
      }
    }
    __syncthreads();

    // contiguous t-major writeback of the residual tile
    {
      const uint4* ls = (const uint4*)xo_lds;
      uint4* gd = (uint4*)(xout + ((size_t)b * T_LEN + t0) * RESC);
      for (int i = tid; i < (TT * RESC) / 8; i += 256) gd[i] = ls[i];
    }
  }
}

// ---------------------------------------------------------------------------
// Output head: relu -> 64x64 conv -> relu -> 64x1 conv (skip is ch-major)
// ---------------------------------------------------------------------------

__global__ __launch_bounds__(256) void final_kernel(
    const float* __restrict__ skip, const float* __restrict__ w1,
    const float* __restrict__ b1, const float* __restrict__ w2,
    const float* __restrict__ b2, float* __restrict__ out) {
  __shared__ float w1s[64 * 64];
  __shared__ float b1s[64];
  __shared__ float w2s[64];
  for (int i = threadIdx.x; i < 64 * 64; i += 256) w1s[i] = w1[i];
  if (threadIdx.x < 64) {
    b1s[threadIdx.x] = b1[threadIdx.x];
    w2s[threadIdx.x] = w2[threadIdx.x];
  }
  __syncthreads();

  const long g = (long)blockIdx.x * 256 + threadIdx.x;
  if (g >= (long)NBATCH * T_LEN) return;
  const int  b = (int)(g / T_LEN);
  const long t = g % T_LEN;
  const float ALPHA = 0.18257418583505536f;  // sqrt(1/30)

  float s[64];
#pragma unroll
  for (int ic = 0; ic < 64; ++ic) {
    const float v = skip[((size_t)(b * 64 + ic)) * T_LEN + t] * ALPHA;
    s[ic] = v > 0.f ? v : 0.f;
  }
  float o = b2[0];
  for (int oc = 0; oc < 64; ++oc) {
    float a = b1s[oc];
#pragma unroll
    for (int ic = 0; ic < 64; ++ic) a += w1s[oc * 64 + ic] * s[ic];
    o += w2s[oc] * (a > 0.f ? a : 0.f);
  }
  out[g] = o;
}

// ---------------------------------------------------------------------------
// Host-side orchestration
// ---------------------------------------------------------------------------

extern "C" void kernel_launch(void* const* d_in, const int* in_sizes, int n_in,
                              void* d_out, int out_size, void* d_ws,
                              size_t ws_size, hipStream_t stream) {
  (void)in_sizes; (void)n_in; (void)out_size; (void)ws_size;

  const float* c          = (const float*)d_in[0];
  const float* noise      = (const float*)d_in[1];
  const float* first_w    = (const float*)d_in[2];
  const float* first_b    = (const float*)d_in[3];
  const float* up_in_w    = (const float*)d_in[4];
  const float* up_conv_w  = (const float*)d_in[5];
  const float* blk_conv_w = (const float*)d_in[6];
  const float* blk_conv_b = (const float*)d_in[7];
  const float* blk_aux_w  = (const float*)d_in[8];
  const float* blk_out_w  = (const float*)d_in[9];
  const float* blk_out_b  = (const float*)d_in[10];
  const float* blk_skip_w = (const float*)d_in[11];
  const float* blk_skip_b = (const float*)d_in[12];
  const float* last1_w    = (const float*)d_in[13];
  const float* last1_b    = (const float*)d_in[14];
  const float* last2_w    = (const float*)d_in[15];
  const float* last2_b    = (const float*)d_in[16];

  char* ws = (char*)d_ws;
  const size_t szXF = (size_t)NBATCH * RESC * T_LEN * sizeof(__bf16); // 39.3 MB
  const size_t szUp = (size_t)NBATCH * AUXC * T_LEN * sizeof(float);  // 98.3 MB
  const size_t szCu = (size_t)NBATCH * AUXP * T_LEN * sizeof(__bf16); // 59.0 MB
  const size_t szC1 = (size_t)NBATCH * AUXC * 300 * sizeof(float);

  // region0: upsample ping buffer, later reused as x0/x1 ping-pong (t-major)
  float*  up_a = (float*)ws;
  __bf16* x0   = (__bf16*)ws;
  __bf16* x1   = (__bf16*)(ws + szXF);
  // region1: upsample pong buffer, later reused as skip accumulator
  float*  up_b = (float*)(ws + szUp);
  float*  skip = (float*)(ws + szUp);
  // region2: upsampled aux features, t-major padded [b][t][96] bf16
  __bf16* cu   = (__bf16*)(ws + 2 * szUp);
  // region3: small pre-upsample conv output
  float*  c1   = (float*)(ws + 2 * szUp + szCu);
  // region4: packed bf16 weights, [30][11][128][32]
  __bf16* wpack = (__bf16*)(ws + 2 * szUp + szCu + ((szC1 + 255) & ~255ul));

  // ---- one-time weight pre-pack (bf16, LDS-slice swizzled) ---------------
  pack_weights_kernel<<<(int)(((long)NBLK * WPACK_E + 255) / 256), 256, 0,
                        stream>>>(blk_conv_w, blk_aux_w, blk_skip_w, blk_out_w,
                                  wpack);

  // ---- upsampling network ------------------------------------------------
  up_in_kernel<<<(NBATCH * AUXC * 300 + 255) / 256, 256, 0, stream>>>(
      c, up_in_w, c1);
  const long NU  = (long)NBATCH * AUXC * T_LEN;
  const int  gUp = (int)((NU + 255) / 256);
  up_conv_kernel<<<gUp, 256, 0, stream>>>(c1,   up_a, up_conv_w + 0,  1);
  up_conv_kernel<<<gUp, 256, 0, stream>>>(up_a, up_b, up_conv_w + 9,  0);
  up_conv_kernel<<<gUp, 256, 0, stream>>>(up_b, up_a, up_conv_w + 18, 0);
  const long NP = (long)NBATCH * T_LEN * AUXP;
  up_conv_pad_kernel<<<(int)((NP + 255) / 256), 256, 0, stream>>>(
      up_a, cu, up_conv_w + 27);

  // ---- first conv + skip zero-init --------------------------------------
  const long NF = (long)NBATCH * T_LEN * RESC;
  first_kernel<<<(int)((NF + 255) / 256), 256, 0, stream>>>(
      noise, first_w, first_b, x0, skip);

  // ---- residual blocks ---------------------------------------------------
  dim3 grid(T_LEN / TT / NT, NBATCH);
  for (int i = 0; i < NBLK; ++i) {
    const int dil = 1 << (i % 10);
    const __bf16* xi = (i & 1) ? x1 : x0;
    __bf16*       xo = (i & 1) ? x0 : x1;
    wavenet_block_kernel<<<grid, 256, 0, stream>>>(
        xi, xo, cu, skip,
        wpack + (size_t)i * WPACK_E,
        blk_conv_b + (size_t)i * GATEC,
        blk_skip_b + (size_t)i * SKIPC,
        blk_out_b + (size_t)i * RESC,
        dil);
  }

  // ---- output head -------------------------------------------------------
  final_kernel<<<(int)(((long)NBATCH * T_LEN + 255) / 256), 256, 0, stream>>>(
      skip, last1_w, last1_b, last2_w, last2_b, (float*)d_out);
}
